// TriangleMultiplicationOutgoing_7456063225918
// MI455X (gfx1250) — matrix operand order repaired and checked by reading, not compiled, not run on hardware
//
#include <hip/hip_runtime.h>
#include <cstdint>

// ---------------- CDNA5 WMMA types & helpers ----------------
typedef __attribute__((ext_vector_type(16))) __bf16 v16bf;
typedef __attribute__((ext_vector_type(8)))  __bf16 v8bf;
typedef __attribute__((ext_vector_type(4)))  __bf16 v4bf;
typedef __attribute__((ext_vector_type(8)))  float  v8f;

#define NN 384
#define CC 128
#define NR (384 * 384)   // 147456 rows

static __device__ __forceinline__ v16bf frag_cat(const __bf16* lo, const __bf16* hi) {
    v8bf a = *(const v8bf*)lo;
    v8bf b = *(const v8bf*)hi;
    v16bf r;
#pragma unroll
    for (int i = 0; i < 8; ++i) { r[i] = a[i]; r[i + 8] = b[i]; }
    return r;
}

static __device__ __forceinline__ v8f wmma_bf16(v16bf a, v16bf b, v8f c) {
    // D = A(16x32) * B(32x16) + C, f32 accumulate
    return __builtin_amdgcn_wmma_f32_16x16x32_bf16(false, a, false, b, (short)0, c, false, false);
}

// A-matrix 16x32 bf16 fragment: lane holds row M = lane&15.
//   elements 0..7  -> K = k0 + 8*(lane>>4) + e
//   elements 8..15 -> K = k0 + 16 + 8*(lane>>4) + (e-8)
static __device__ __forceinline__ v16bf load_A(const __bf16* rowmaj, int rowBase,
                                               int stride, int k0, int lane) {
    const __bf16* p = rowmaj + (size_t)(rowBase + (lane & 15)) * stride + k0 + 8 * (lane >> 4);
    return frag_cat(p, p + 16);
}

// B-matrix 32x16 bf16 fragment: lane holds column N = lane&15.
//   elements 0..7  -> K = k0 + 8*(lane>>4) + e
//   elements 8..15 -> K = k0 + 16 + 8*(lane>>4) + (e-8)
// Source layout must be [N][K] row-major (K contiguous).
static __device__ __forceinline__ v16bf load_B(const __bf16* nk, int nBase,
                                               int stride, int k0, int lane) {
    const __bf16* p = nk + (size_t)(nBase + (lane & 15)) * stride + k0 + 8 * (lane >> 4);
    return frag_cat(p, p + 16);
}

// Fast sigmoid: v_exp_f32 + v_rcp_f32 (no IEEE divide expansion).
static __device__ __forceinline__ float sigmoidf(float x) {
    return __builtin_amdgcn_rcpf(1.0f + __expf(-x));
}

// ---------------- Kernel 0: weights f32 -> bf16 ----------------
__global__ __launch_bounds__(256) void k_cvt_weights(
    const float* __restrict__ w0, const float* __restrict__ w1,
    const float* __restrict__ w2, const float* __restrict__ w3,
    const float* __restrict__ w4, const float* __restrict__ w5,
    __bf16* __restrict__ dst) {
    int i = blockIdx.x * 256 + threadIdx.x;   // 0 .. 6*16384-1
    int m = i >> 14;
    int off = i & 16383;
    float v;
    if      (m == 0) v = w0[off];
    else if (m == 1) v = w1[off];
    else if (m == 2) v = w2[off];
    else if (m == 3) v = w3[off];
    else if (m == 4) v = w4[off];
    else             v = w5[off];
    dst[i] = (__bf16)v;
}

// ---------------- Kernel 1: fused LayerNorm + 5 projections ----------------
// Block = 128 threads (4 waves). Each wave owns a 16-row M-tile.
// Outputs: a_t,b_t channel-major bf16 [d][row], g row-major f32 [row][d].
__global__ __launch_bounds__(128) void k_ln_proj(
    const float* __restrict__ z,
    const float* __restrict__ lnin_g, const float* __restrict__ lnin_b,
    const __bf16* __restrict__ wbf,    // [6][128][128]: Wa1,Wa2,Wb1,Wb2,Wg,Wout
    __bf16* __restrict__ a_t, __bf16* __restrict__ b_t,
    float* __restrict__ g_out) {
    __shared__ __align__(16) __bf16 zn[4][16][CC];

    const int lane = threadIdx.x & 31;
    const int wave = threadIdx.x >> 5;
    const int half = lane >> 4;
    const int r0 = blockIdx.x * 64 + wave * 16;

    // ---- LayerNorm of 16 rows (wave-cooperative, 4 f32 per lane per row) ----
    const float4 gg = ((const float4*)lnin_g)[lane];
    const float4 bb = ((const float4*)lnin_b)[lane];
    for (int rr = 0; rr < 16; ++rr) {
        const int row = r0 + rr;
        float4 v = ((const float4*)(z + (size_t)row * CC))[lane];
        float s = v.x + v.y + v.z + v.w;
        float q = v.x * v.x + v.y * v.y + v.z * v.z + v.w * v.w;
#pragma unroll
        for (int off = 16; off; off >>= 1) {
            s += __shfl_xor(s, off, 32);
            q += __shfl_xor(q, off, 32);
        }
        const float mu = s * (1.0f / 128.0f);
        const float var = q * (1.0f / 128.0f) - mu * mu;
        const float rstd = rsqrtf(var + 1e-5f);
        v4bf pk;
        pk[0] = (__bf16)((v.x - mu) * rstd * gg.x + bb.x);
        pk[1] = (__bf16)((v.y - mu) * rstd * gg.y + bb.y);
        pk[2] = (__bf16)((v.z - mu) * rstd * gg.z + bb.z);
        pk[3] = (__bf16)((v.w - mu) * rstd * gg.w + bb.w);
        *(v4bf*)&zn[wave][rr][lane * 4] = pk;   // one 8-byte LDS store
    }
    __syncthreads();

    const __bf16* zt = &zn[wave][0][0];   // 16 x 128 bf16, stride 128

    // ---- a = sigmoid(z@Wa1^T) * (z@Wa2^T);  b likewise with Wb1/Wb2 ----
    for (int pass = 0; pass < 2; ++pass) {
        const __bf16* W1 = wbf + (size_t)(pass * 2 + 0) * CC * CC;
        const __bf16* W2 = wbf + (size_t)(pass * 2 + 1) * CC * CC;
        __bf16* outp = (pass == 0) ? a_t : b_t;
        for (int nt = 0; nt < 8; ++nt) {
            v8f acc1 = {};
            v8f acc2 = {};
#pragma unroll
            for (int kt = 0; kt < 4; ++kt) {
                const int k0 = kt * 32;
                v16bf af = load_A(zt, 0, CC, k0, lane);
                acc1 = wmma_bf16(af, load_B(W1, nt * 16, CC, k0, lane), acc1);
                acc2 = wmma_bf16(af, load_B(W2, nt * 16, CC, k0, lane), acc2);
            }
            const int d = nt * 16 + (lane & 15);
#pragma unroll
            for (int r = 0; r < 8; ++r) {
                const int grow = r0 + r + 8 * half;
                const float val = sigmoidf(acc1[r]) * acc2[r];
                outp[(size_t)d * NR + grow] = (__bf16)val;   // channel-major scatter
            }
        }
    }

    // ---- g = sigmoid(z@Wg^T), stored row-major f32 ----
    const __bf16* Wg = wbf + (size_t)4 * CC * CC;
    for (int nt = 0; nt < 8; ++nt) {
        v8f acc = {};
#pragma unroll
        for (int kt = 0; kt < 4; ++kt) {
            const int k0 = kt * 32;
            acc = wmma_bf16(load_A(zt, 0, CC, k0, lane),
                            load_B(Wg, nt * 16, CC, k0, lane), acc);
        }
        const int d = nt * 16 + (lane & 15);
#pragma unroll
        for (int r = 0; r < 8; ++r) {
            const int grow = r0 + r + 8 * half;
            g_out[(size_t)grow * CC + d] = sigmoidf(acc[r]);
        }
    }
}

// ---------------- Kernel 2: einsum + LN(s) + @Wout^T + gate ----------------
// Block = 256 threads (8 waves). Tile = 16 i-rows x 16 j-cols (256 pairs).
// Phase A: 128 per-channel 16x16x384 GEMMs (one channel per wave per step).
// Phase B: LayerNorm over d per pair in LDS (vectorized float4 reads).
// Phase C: [256 x 128] x [128 x 128] Wout GEMM + sigmoid gate -> out.
#define SPITCH 132
__global__ __launch_bounds__(256) void k_einsum_out(
    const __bf16* __restrict__ a_t, const __bf16* __restrict__ b_t,
    const float* __restrict__ g_in,
    const float* __restrict__ lnout_g, const float* __restrict__ lnout_b,
    const __bf16* __restrict__ woutbf,
    float* __restrict__ out) {
    __shared__ float s_lds[256 * SPITCH];                 // s[pair][d], f32
    __shared__ __align__(16) __bf16 sn[256 * CC];         // LN(s) in bf16

    const int lane = threadIdx.x & 31;
    const int wave = threadIdx.x >> 5;
    const int half = lane >> 4;
    const int i0 = blockIdx.x * 16;
    const int j0 = blockIdx.y * 16;

    // ---- Phase A: s[i,j,d] = sum_k a[i,k,d]*b[j,k,d] (C = A * B^T per d) ----
    for (int db = 0; db < 16; ++db) {
        const int d = db * 8 + wave;
        const __bf16* aP = a_t + (size_t)d * NR;
        const __bf16* bP = b_t + (size_t)d * NR;
        v8f acc = {};
#pragma unroll
        for (int kt = 0; kt < 12; ++kt) {
            const int k0 = kt * 32;
            v16bf af = load_A(aP, i0, NN, k0, lane);       // A[m][k] = a[i0+m][k]
            v16bf bf = load_B(bP, j0, NN, k0, lane);       // B[k][n] = b[j0+n][k]
            acc = wmma_bf16(af, bf, acc);
        }
#pragma unroll
        for (int r = 0; r < 8; ++r) {
            const int pair = (r + 8 * half) * 16 + (lane & 15);
            s_lds[pair * SPITCH + d] = acc[r];
        }
    }
    __syncthreads();

    // ---- Phase B: LayerNorm over d (one thread per pair, float4 vectorized) ----
    {
        const int t = threadIdx.x;
        const float4* srow4 = (const float4*)&s_lds[t * SPITCH];  // 528B offset, 16B aligned
        float s = 0.f, q = 0.f;
#pragma unroll
        for (int d4 = 0; d4 < 32; ++d4) {
            const float4 v = srow4[d4];
            s += v.x + v.y + v.z + v.w;
            q += v.x * v.x + v.y * v.y + v.z * v.z + v.w * v.w;
        }
        const float mu = s * (1.0f / 128.0f);
        const float var = q * (1.0f / 128.0f) - mu * mu;
        const float rstd = rsqrtf(var + 1e-5f);
        const float4* g4 = (const float4*)lnout_g;
        const float4* b4 = (const float4*)lnout_b;
#pragma unroll
        for (int d4 = 0; d4 < 32; ++d4) {
            const float4 v = srow4[d4];
            const float4 gv = g4[d4];
            const float4 bv = b4[d4];
            v4bf pk;
            pk[0] = (__bf16)((v.x - mu) * rstd * gv.x + bv.x);
            pk[1] = (__bf16)((v.y - mu) * rstd * gv.y + bv.y);
            pk[2] = (__bf16)((v.z - mu) * rstd * gv.z + bv.z);
            pk[3] = (__bf16)((v.w - mu) * rstd * gv.w + bv.w);
            *(v4bf*)&sn[t * CC + d4 * 4] = pk;   // one 8-byte LDS store
        }
    }
    __syncthreads();

    // ---- Phase C: out = g * (LN(s) @ Wout^T). M=256 pairs, N=128, K=128 ----
    for (int t = 0; t < 16; ++t) {
        const int tile = wave * 16 + t;     // 128 (mt,nt) tiles over 8 waves
        const int mt = tile >> 3;
        const int nt = tile & 7;
        v8f acc = {};
#pragma unroll
        for (int kt = 0; kt < 4; ++kt) {
            const int k0 = kt * 32;
            v16bf af = load_A(sn, mt * 16, CC, k0, lane);
            v16bf bf = load_B(woutbf, nt * 16, CC, k0, lane);
            acc = wmma_bf16(af, bf, acc);
        }
        const int dch = nt * 16 + (lane & 15);
#pragma unroll
        for (int r = 0; r < 8; ++r) {
            const int pair = mt * 16 + r + 8 * half;
            const int li = pair >> 4;
            const int lj = pair & 15;
            const size_t row = (size_t)(i0 + li) * NN + (j0 + lj);
            const size_t idx = row * CC + dch;
            out[idx] = g_in[idx] * acc[r];
        }
    }
}

// ---------------- Host launcher ----------------
extern "C" void kernel_launch(void* const* d_in, const int* in_sizes, int n_in,
                              void* d_out, int out_size, void* d_ws, size_t ws_size,
                              hipStream_t stream) {
    const float* z       = (const float*)d_in[0];
    const float* lnin_g  = (const float*)d_in[1];
    const float* lnin_b  = (const float*)d_in[2];
    const float* lnout_g = (const float*)d_in[3];
    const float* lnout_b = (const float*)d_in[4];
    const float* Wa1     = (const float*)d_in[5];
    const float* Wa2     = (const float*)d_in[6];
    const float* Wb1     = (const float*)d_in[7];
    const float* Wb2     = (const float*)d_in[8];
    const float* Wg      = (const float*)d_in[9];
    const float* Wout    = (const float*)d_in[10];

    char* ws = (char*)d_ws;
    // ws layout (bytes):
    //   [0)            wbf:  6*128*128 bf16  = 196608
    //   [196608)       a_t:  128*147456 bf16 = 37748736
    //   [37945344)     b_t:  128*147456 bf16 = 37748736
    //   [75694080)     g  :  147456*128 f32  = 75497472   (total ~144 MB)
    __bf16* wbf = (__bf16*)(ws);
    __bf16* a_t = (__bf16*)(ws + 196608);
    __bf16* b_t = (__bf16*)(ws + 37945344);
    float*  g_w = (float*)(ws + 75694080);
    float*  out = (float*)d_out;

    k_cvt_weights<<<384, 256, 0, stream>>>(Wa1, Wa2, Wb1, Wb2, Wg, Wout, wbf);

    // 147456 rows / 64 rows per block
    k_ln_proj<<<2304, 128, 0, stream>>>(z, lnin_g, lnin_b, wbf, a_t, b_t, g_w);

    // 24 x 24 tiles of 16x16 (i,j) pairs
    k_einsum_out<<<dim3(24, 24), 256, 0, stream>>>(
        a_t, b_t, g_w, lnout_g, lnout_b, wbf + (size_t)5 * 128 * 128, out);
}
